// PeriodicSetTransformerEncoder_45337674776589
// MI455X (gfx1250) — compile-verified, matrix operand order checked by reading, not run on hardware
//
#include <hip/hip_runtime.h>
#include <hip/hip_bf16.h>

// B=16, N=D=128, fp32 in/out. bf16 WMMA (f32 accum) for all GEMMs.
// Operand staging via GLOBAL_LOAD_ASYNC_TO_LDS_B128 (ASYNCcnt).

typedef __attribute__((ext_vector_type(16))) __bf16 v16bf;
typedef __attribute__((ext_vector_type(8)))  __bf16 v8bf;
typedef __attribute__((ext_vector_type(8)))  float  v8f;

#define EPSF 1e-5f

__device__ __forceinline__ float mishf(float x) {
    float sp = logf(1.0f + __expf(x));
    return x * tanhf(sp);
}

// Async copy of 16 bytes/lane: global (saddr base + 32-bit voffset) -> LDS byte
// address `lds_off` (dynamic LDS starts at offset 0; these kernels use only
// dynamic LDS). Tracked by ASYNCcnt.
__device__ __forceinline__ void async_copy_b128(int lds_off, int goff, const void* gbase) {
    asm volatile("global_load_async_to_lds_b128 %0, %1, %2"
                 :: "v"(lds_off), "v"(goff), "s"(gbase) : "memory");
}
__device__ __forceinline__ void async_wait0() {
    asm volatile("s_wait_asynccnt 0x0" ::: "memory");
}

// Build a 16x32 bf16 WMMA fragment from row-major LDS (row stride 128 bf16).
// ISA 7.12.2: lanes 0-15 hold K 0..7 (elems 0-7) and K 16..23 (elems 8-15);
// lanes 16-31 hold K 8..15 and K 24..31. Caller passes base = &row[kbase + half*8].
__device__ __forceinline__ v16bf load_frag(const __bf16* base) {
    v8bf lo = *reinterpret_cast<const v8bf*>(base);        // K +0..7
    v8bf hi = *reinterpret_cast<const v8bf*>(base + 16);   // K +16..23
    return __builtin_shufflevector(lo, hi, 0,1,2,3,4,5,6,7,8,9,10,11,12,13,14,15);
}

// 128x128x128 GEMM core: A_lds row-major [128][128] bf16, Bt_lds = W^T [N][K] bf16.
// Wave `wave` produces rows [16w,16w+16), all 128 cols as 8 tiles of 16x16 (f32 acc).
__device__ __forceinline__ void mma128(const __bf16* A_lds, const __bf16* Bt_lds,
                                       int wave, int l16, int half, v8f acc[8]) {
#pragma unroll
    for (int t = 0; t < 8; ++t)
#pragma unroll
        for (int e = 0; e < 8; ++e) acc[t][e] = 0.0f;
#pragma unroll
    for (int kk = 0; kk < 4; ++kk) {
        const __bf16* ap = A_lds + (wave * 16 + l16) * 128 + kk * 32 + half * 8;
        v16bf a = load_frag(ap);
#pragma unroll
        for (int t = 0; t < 8; ++t) {
            const __bf16* bp = Bt_lds + (t * 16 + l16) * 128 + kk * 32 + half * 8;
            v16bf b = load_frag(bp);
            acc[t] = __builtin_amdgcn_wmma_f32_16x16x32_bf16(
                false, a, false, b, (short)0, acc[t], false, false);
        }
    }
}

// ---------------------------------------------------------------------------
// One-time prep: bf16 transposed copies of the 7 weight matrices.
// dst[m][n*128+k] = (bf16) W_m[k*128+n]
__global__ void convert_weights_kernel(const float* __restrict__ W0, const float* __restrict__ W1,
                                       const float* __restrict__ W2, const float* __restrict__ W3,
                                       const float* __restrict__ W4, const float* __restrict__ W5,
                                       const float* __restrict__ W6, __bf16* __restrict__ dst) {
    const float* Ws[7] = {W0, W1, W2, W3, W4, W5, W6};
    const float* W = Ws[blockIdx.x];
    __bf16* d = dst + (long)blockIdx.x * 128 * 128;
    for (int idx = threadIdx.x; idx < 128 * 128; idx += 256) {
        int k = idx >> 7, n = idx & 127;
        d[n * 128 + k] = (__bf16)W[idx];
    }
}

// ---------------------------------------------------------------------------
// Row-wise LayerNorm over 128 cols, one wave32 per row (4 elems/lane).
// Optional addend X2, nullable f32/bf16 outputs, optional transposed store.
__global__ void ln_rows_kernel(const float* __restrict__ X, const float* __restrict__ X2,
                               const float* __restrict__ g, const float* __restrict__ bb,
                               float* __restrict__ out, __bf16* __restrict__ out_bf,
                               int do_transpose) {
    const int tid = threadIdx.x;
    const int wave = tid >> 5, lane = tid & 31;
    const long r = (long)blockIdx.x * 8 + wave;  // row index over B*N = 2048
    float vals[4];
    float s1 = 0.f, s2 = 0.f;
#pragma unroll
    for (int e = 0; e < 4; ++e) {
        int c = lane + 32 * e;
        float x = X[r * 128 + c];
        if (X2) x += X2[r * 128 + c];
        vals[e] = x; s1 += x; s2 += x * x;
    }
#pragma unroll
    for (int m = 1; m <= 16; m <<= 1) {
        s1 += __shfl_xor(s1, m, 32);
        s2 += __shfl_xor(s2, m, 32);
    }
    float mean = s1 * (1.0f / 128.0f);
    float var  = s2 * (1.0f / 128.0f) - mean * mean;
    float is   = rsqrtf(var + EPSF);
#pragma unroll
    for (int e = 0; e < 4; ++e) {
        int c = lane + 32 * e;
        float y = (vals[e] - mean) * is * g[c] + bb[c];
        long o;
        if (do_transpose) {  // feat[b, c, n] = xn[b, n, c]
            long b = r >> 7, n = r & 127;
            o = (b * 128 + c) * 128 + n;
        } else {
            o = r * 128 + c;
        }
        if (out)    out[o]    = y;
        if (out_bf) out_bf[o] = (__bf16)y;
    }
}

// ---------------------------------------------------------------------------
// Generic 128-col GEMM: out[row,:] = epi( A[row,:] @ W + bias ), 128 rows/block.
// A and W^T are bf16 in global, async-copied straight into LDS.
// flags: 1 = row LayerNorm, 2 = mish, 4 = += residual, 8 = has bias.
#define GF_LN   1
#define GF_MISH 2
#define GF_RES  4
#define GF_BIAS 8

__global__ void gemm128_kernel(const __bf16* __restrict__ A_bf, const __bf16* __restrict__ Wt_bf,
                               const float* __restrict__ bias,
                               const float* __restrict__ ln_g, const float* __restrict__ ln_b,
                               const float* __restrict__ residual,
                               float* __restrict__ out, __bf16* __restrict__ out_bf, int flags) {
    extern __shared__ char smem[];
    const __bf16* A_lds  = reinterpret_cast<const __bf16*>(smem);  // bytes [0, 32K)
    const __bf16* Bt_lds = A_lds + 128 * 128;                      // bytes [32K, 64K)
    const int tid = threadIdx.x;
    const long abase = (long)blockIdx.x * 128 * 128;

    // Async-stage A tile (32 KB) and W^T (32 KB) into LDS: 16 B/lane x 8 iters.
    {
        const __bf16* ag = A_bf + abase;
#pragma unroll
        for (int it = 0; it < 8; ++it) {
            int off = it * 4096 + tid * 16;
            async_copy_b128(off,         off, (const void*)ag);
            async_copy_b128(32768 + off, off, (const void*)Wt_bf);
        }
        async_wait0();
    }
    __syncthreads();

    const int wave = tid >> 5, lane = tid & 31;
    const int half = lane >> 4, l16 = lane & 15;
    v8f acc[8];
    mma128(A_lds, Bt_lds, wave, l16, half, acc);

    if (flags & GF_BIAS) {
#pragma unroll
        for (int t = 0; t < 8; ++t) {
            float bv = bias[t * 16 + l16];
#pragma unroll
            for (int e = 0; e < 8; ++e) acc[t][e] += bv;
        }
    }
    if (flags & GF_RES) {
#pragma unroll
        for (int t = 0; t < 8; ++t)
#pragma unroll
            for (int e = 0; e < 8; ++e) {
                int row = wave * 16 + e + 8 * half;
                int col = t * 16 + l16;
                acc[t][e] += residual[abase + (long)row * 128 + col];
            }
    }
    if (flags & GF_LN) {
        // per-row stats; xor masks 1/2/4/8 keep the two half-wave row groups
        // (rows e vs e+8) reducing independently.
#pragma unroll
        for (int e = 0; e < 8; ++e) {
            float s1 = 0.f, s2 = 0.f;
#pragma unroll
            for (int t = 0; t < 8; ++t) { float x = acc[t][e]; s1 += x; s2 += x * x; }
#pragma unroll
            for (int m = 1; m <= 8; m <<= 1) {
                s1 += __shfl_xor(s1, m, 32);
                s2 += __shfl_xor(s2, m, 32);
            }
            float mean = s1 * (1.0f / 128.0f);
            float var  = s2 * (1.0f / 128.0f) - mean * mean;
            float is   = rsqrtf(var + EPSF);
#pragma unroll
            for (int t = 0; t < 8; ++t) {
                int col = t * 16 + l16;
                acc[t][e] = (acc[t][e] - mean) * is * ln_g[col] + ln_b[col];
            }
        }
    }
    if (flags & GF_MISH) {
#pragma unroll
        for (int t = 0; t < 8; ++t)
#pragma unroll
            for (int e = 0; e < 8; ++e) acc[t][e] = mishf(acc[t][e]);
    }
#pragma unroll
    for (int t = 0; t < 8; ++t)
#pragma unroll
        for (int e = 0; e < 8; ++e) {
            int row = wave * 16 + e + 8 * half;
            int col = t * 16 + l16;
            long o = abase + (long)row * 128 + col;
            if (out)    out[o]    = acc[t][e];
            if (out_bf) out_bf[o] = (__bf16)acc[t][e];
        }
}

// ---------------------------------------------------------------------------
// Fused relation-attention: one block per (b, i).
//   u[j,m] = mish(LN_m(kW1[b,j,m] - qW1[b,i,m]))      (be1 folded into kW1)
//   w[j,n] = (u @ We2)[j,n] + be2[n]                    (WMMA in LDS)
//   weighted softmax over j, mask, contract with v ->  att_bf[b, n, i]
__global__ void attn_kernel(const float* __restrict__ kW1, const float* __restrict__ qW1,
                            const __bf16* __restrict__ We2t_bf, const float* __restrict__ be2,
                            const float* __restrict__ lne_g, const float* __restrict__ lne_b,
                            const float* __restrict__ V, const float* __restrict__ wts,
                            __bf16* __restrict__ att_bf) {
    extern __shared__ char smem[];
    __bf16* u_lds = reinterpret_cast<__bf16*>(smem);            // bytes [0, 32K)
    __bf16* b_lds = u_lds + 128 * 128;                          // bytes [32K, 64K)
    float*  w_lds = reinterpret_cast<float*>(smem + 65536);     // bytes [64K, 128K)
    const int tid = threadIdx.x;
    const int b = blockIdx.x >> 7;
    const int i = blockIdx.x & 127;

    // Kick off async We2^T -> LDS copy; overlaps with phase-1 VALU work.
#pragma unroll
    for (int it = 0; it < 8; ++it) {
        int off = it * 4096 + tid * 16;
        async_copy_b128(32768 + off, off, (const void*)We2t_bf);
    }

    // Phase 1: u rows. 2 threads per row j, pair-reductions via shfl_xor(1).
    {
        const int j = tid >> 1, h = tid & 1;
        const float* kr = kW1 + ((long)b * 128 + j) * 128;
        const float* qr = qW1 + ((long)b * 128 + i) * 128;
        float s1 = 0.f, s2 = 0.f;
        for (int e = 0; e < 64; ++e) {
            int m = h * 64 + e;
            float val = kr[m] - qr[m];        // be1 already in kW1
            w_lds[j * 128 + m] = val;         // stash pre-LN value (scratch)
            s1 += val; s2 += val * val;
        }
        s1 += __shfl_xor(s1, 1, 32);
        s2 += __shfl_xor(s2, 1, 32);
        float mean = s1 * (1.0f / 128.0f);
        float var  = s2 * (1.0f / 128.0f) - mean * mean;
        float is   = rsqrtf(var + EPSF);
        for (int e = 0; e < 64; ++e) {
            int m = h * 64 + e;
            float val = (w_lds[j * 128 + m] - mean) * is * lne_g[m] + lne_b[m];
            u_lds[j * 128 + m] = (__bf16)mishf(val);
        }
    }
    async_wait0();
    __syncthreads();

    // Phase 2: w = u @ We2 + be2 (WMMA), result to LDS.
    {
        const int wave = tid >> 5, lane = tid & 31;
        const int half = lane >> 4, l16 = lane & 15;
        v8f acc[8];
        mma128(u_lds, b_lds, wave, l16, half, acc);
#pragma unroll
        for (int t = 0; t < 8; ++t) {
            float bv = be2[t * 16 + l16];
#pragma unroll
            for (int e = 0; e < 8; ++e) {
                int row = wave * 16 + e + 8 * half;
                w_lds[row * 128 + t * 16 + l16] = acc[t][e] + bv;
            }
        }
    }
    __syncthreads();

    // Phase 3: weighted softmax over j + mask + contraction with v, per column n.
    {
        const int col = tid >> 1, h = tid & 1;
        const float wtI = wts[b * 128 + i];
        float mx = -3.0e38f;
        for (int e = 0; e < 64; ++e) {
            int j = h * 64 + e;
            mx = fmaxf(mx, w_lds[j * 128 + col]);
        }
        mx = fmaxf(mx, __shfl_xor(mx, 1, 32));
        float s = 0.f, o = 0.f;
        for (int e = 0; e < 64; ++e) {
            int j = h * 64 + e;
            float wj = wts[b * 128 + j];
            float ex = __expf(w_lds[j * 128 + col] - mx) * wj;
            s += ex;
            if (wtI * wj > 0.f)
                o += ex * V[((long)b * 128 + j) * 128 + col];
        }
        s += __shfl_xor(s, 1, 32);
        o += __shfl_xor(o, 1, 32);
        if (h == 0)
            att_bf[((long)b * 128 + col) * 128 + i] = (__bf16)(o / s);  // att[b, n, i]
    }
}

// ---------------------------------------------------------------------------
extern "C" void kernel_launch(void* const* d_in, const int* in_sizes, int n_in,
                              void* d_out, int out_size, void* d_ws, size_t ws_size,
                              hipStream_t stream) {
    (void)in_sizes; (void)n_in; (void)out_size; (void)ws_size;
    const float* x     = (const float*)d_in[0];
    const float* wts   = (const float*)d_in[1];
    const float* ln_g  = (const float*)d_in[2];
    const float* ln_b  = (const float*)d_in[3];
    const float* Wq    = (const float*)d_in[4];
    const float* bq    = (const float*)d_in[5];
    const float* lnq_g = (const float*)d_in[6];
    const float* lnq_b = (const float*)d_in[7];
    const float* Wk    = (const float*)d_in[8];
    const float* bk    = (const float*)d_in[9];
    const float* lnk_g = (const float*)d_in[10];
    const float* lnk_b = (const float*)d_in[11];
    const float* Wv    = (const float*)d_in[12];
    const float* bv    = (const float*)d_in[13];
    const float* We1   = (const float*)d_in[14];
    const float* be1   = (const float*)d_in[15];
    const float* lne_g = (const float*)d_in[16];
    const float* lne_b = (const float*)d_in[17];
    const float* We2   = (const float*)d_in[18];
    const float* be2   = (const float*)d_in[19];
    const float* Wout  = (const float*)d_in[20];
    const float* bout  = (const float*)d_in[21];
    const float* Wffn  = (const float*)d_in[22];
    const float* bffn  = (const float*)d_in[23];
    float* out = (float*)d_out;

    const long S = 16L * 128 * 128;  // 262144 elems per [B,N,D] buffer
    float* ws   = (float*)d_ws;
    // f32 buffers
    float* v    = ws + 0 * S;
    float* qW1  = ws + 1 * S;
    float* kW1  = ws + 2 * S;
    float* out1 = ws + 3 * S;
    float* out2 = ws + 4 * S;
    // bf16 buffers
    __bf16* bfb     = (__bf16*)(ws + 5 * S);
    __bf16* feat_bf = bfb + 0 * S;
    __bf16* q_bf    = bfb + 1 * S;
    __bf16* k_bf    = bfb + 2 * S;
    __bf16* att_bf  = bfb + 3 * S;
    __bf16* ln1_bf  = bfb + 4 * S;
    __bf16* wt_bf   = bfb + 5 * S;  // 7 x [128][128] transposed bf16 weights
    const long WT = 128 * 128;
    __bf16* WqT = wt_bf + 0 * WT, *WkT = wt_bf + 1 * WT, *WvT = wt_bf + 2 * WT;
    __bf16* We1T = wt_bf + 3 * WT, *We2T = wt_bf + 4 * WT;
    __bf16* WoutT = wt_bf + 5 * WT, *WffnT = wt_bf + 6 * WT;

    const size_t gemm_smem = 2 * 128 * 128 * sizeof(__bf16);            // 64 KB
    const size_t attn_smem = gemm_smem + 128 * 128 * sizeof(float);     // 128 KB
    hipFuncSetAttribute((const void*)gemm128_kernel,
                        hipFuncAttributeMaxDynamicSharedMemorySize, (int)gemm_smem);
    hipFuncSetAttribute((const void*)attn_kernel,
                        hipFuncAttributeMaxDynamicSharedMemorySize, (int)attn_smem);

    // 0. bf16-transpose all weights once
    convert_weights_kernel<<<7, 256, 0, stream>>>(Wq, Wk, Wv, We1, We2, Wout, Wffn, wt_bf);
    // 1. feat_bf[b,d,n] = LN(x[b,n,:])[d]   (transposed store)
    ln_rows_kernel<<<256, 256, 0, stream>>>(x, nullptr, ln_g, ln_b, nullptr, feat_bf, 1);
    // 2-4. q / k / v
    gemm128_kernel<<<16, 256, gemm_smem, stream>>>(feat_bf, WqT, bq, lnq_g, lnq_b, nullptr,
                                                   nullptr, q_bf, GF_BIAS | GF_LN | GF_MISH);
    gemm128_kernel<<<16, 256, gemm_smem, stream>>>(feat_bf, WkT, bk, lnk_g, lnk_b, nullptr,
                                                   nullptr, k_bf, GF_BIAS | GF_LN | GF_MISH);
    gemm128_kernel<<<16, 256, gemm_smem, stream>>>(feat_bf, WvT, bv, nullptr, nullptr, nullptr,
                                                   v, nullptr, GF_BIAS);
    // 5-6. rel@We1 decomposed: qW1 = q@We1 ; kW1 = k@We1 + be1
    gemm128_kernel<<<16, 256, gemm_smem, stream>>>(q_bf, We1T, nullptr, nullptr, nullptr,
                                                   nullptr, qW1, nullptr, 0);
    gemm128_kernel<<<16, 256, gemm_smem, stream>>>(k_bf, We1T, be1, nullptr, nullptr,
                                                   nullptr, kW1, nullptr, GF_BIAS);
    // 7. fused relation attention -> att_bf[B,N,D]
    attn_kernel<<<16 * 128, 256, attn_smem, stream>>>(kW1, qW1, We2T, be2, lne_g, lne_b,
                                                      v, wts, att_bf);
    // 8. out1 = x + att@Wout + bout
    gemm128_kernel<<<16, 256, gemm_smem, stream>>>(att_bf, WoutT, bout, nullptr, nullptr, x,
                                                   out1, nullptr, GF_BIAS | GF_RES);
    // 9-10. out2 = mish(LN(out1)@Wffn + bffn)
    ln_rows_kernel<<<256, 256, 0, stream>>>(out1, nullptr, ln_g, ln_b, nullptr, ln1_bf, 0);
    gemm128_kernel<<<16, 256, gemm_smem, stream>>>(ln1_bf, WffnT, bffn, nullptr, nullptr,
                                                   nullptr, out2, nullptr, GF_BIAS | GF_MISH);
    // 11. out = LN(out1 + out2)
    ln_rows_kernel<<<256, 256, 0, stream>>>(out1, out2, ln_g, ln_b, out, nullptr, 0);
}
